// Encoder_2551210574204
// MI455X (gfx1250) — compile-verified, matrix operand order checked by reading, not run on hardware
//
#include <hip/hip_runtime.h>
#include <cmath>

#define C_   192
#define FC_  768
#define NH   2
#define NL   6
#define WSZ  4
#define BB   8
#define TT   1024
#define DKH  96
#define BHD  (BB*NH)

typedef __bf16 bhalf;
typedef __attribute__((ext_vector_type(4)))  __bf16 v4bf;
typedef __attribute__((ext_vector_type(8)))  __bf16 v8bf;
typedef __attribute__((ext_vector_type(16))) __bf16 v16bf;
typedef __attribute__((ext_vector_type(8)))  float  v8f;
typedef __attribute__((ext_vector_type(4), aligned(4)))  float v4fu;  // dword-aligned vec load

#define BMT 64
#define BNT 128
#define BKT 32
#define LDK 40     // 32 + 8 pad bf16 (80B rows: 16B aligned pitch)
#define NT  256    // 8 waves: 2(m) x 4(n), each wave 32x32 out = 4 WMMA/k-step

__device__ __forceinline__ bhalf f2bf(float f) { return (bhalf)f; }          // native cvt
__device__ __forceinline__ float bf2f(bhalf h) { return (float)h; }
__device__ __forceinline__ v4bf  pack4v(v4fu f) { return __builtin_convertvector(f, v4bf); }

// WMMA 16x16 bf16 fragment from LDS: lane holds K = kh+0..7 and kh+16..23 of its row.
__device__ __forceinline__ v16bf frag_ld(const bhalf* row, int kh) {
  v8bf lo = *(const v8bf*)(row + kh);
  v8bf hi = *(const v8bf*)(row + kh + 16);
  return __builtin_shufflevector(lo, hi, 0,1,2,3,4,5,6,7,8,9,10,11,12,13,14,15);
}
__device__ __forceinline__ v8f wmma_bf(v16bf a, v16bf b, v8f c) {
  return __builtin_amdgcn_wmma_f32_16x16x32_bf16(false, a, false, b, (short)0, c, false, false);
}
__device__ __forceinline__ unsigned lds_addr(const void* p) {
  return (unsigned)(uintptr_t)p;   // low 32 bits of generic addr = LDS byte offset
}

// ---------------------------------------------------------------------------
// Generic conv1d-as-GEMM: Y[b,m,t] = act( sum_{c,tap} W[m,c*R+tap]*X[b,c,t+tap-(R-1)/2] + bias[m] )
template<int R, bool RELU, bool MIN_, bool MOUT>
__global__ __launch_bounds__(NT) void gemm_conv_kernel(
    const float* __restrict__ Wt, const float* __restrict__ bias,
    const float* __restrict__ X, const float* __restrict__ xmask,
    float* __restrict__ Y, int M, int Kc, float alpha)
{
  __shared__ alignas(16) bhalf Al[BMT*LDK];
  __shared__ alignas(16) bhalf Bl[BNT*LDK];
  const int b  = blockIdx.z;
  const int m0 = blockIdx.y * BMT;
  const int n0 = blockIdx.x * BNT;
  const int Ktot = Kc * R;
  const int tid = threadIdx.x, lane = tid & 31, wid = tid >> 5;
  const int wm = wid & 1, wn = wid >> 1;
  const int fr = lane & 15, kh = (lane >> 4) << 3;
  v8f a00 = {0,0,0,0,0,0,0,0}, a01 = a00, a10 = a00, a11 = a00;

  for (int k0 = 0; k0 < Ktot; k0 += BKT) {
    if (k0 + BKT < Ktot)  // stream next weight block (global_prefetch_b8)
      __builtin_prefetch(&Wt[(size_t)(m0 + (tid & 63)) * Ktot + k0 + BKT], 0, 1);
    // ---- A: weights, contiguous K both in global and LDS: b128 load + b64 store
    #pragma unroll
    for (int it = 0; it < (BMT*BKT)/(4*NT); ++it) {
      int g = tid + it*NT;                 // 512 float4 groups
      int k4 = (g & 7) * 4;
      int m  = g >> 3;
      int gm = m0 + m;
      v4fu w = {0,0,0,0};
      if (gm < M) w = *(const v4fu*)&Wt[(size_t)gm * Ktot + k0 + k4];
      *(v4bf*)&Al[m*LDK + k4] = pack4v(w);
    }
    // ---- B: contiguous along t(n) in global, along k in LDS: 4x4 register transpose
    {
      int bn = (tid & 31) * 4;             // 32 n-blocks
      int bk = (tid >> 5) * 4;             // 8 k-blocks
      v4fu f[4];
      #pragma unroll
      for (int j = 0; j < 4; ++j) {
        int gk = k0 + bk + j;
        int c = gk / R, tap = gk - c*R;
        const float* src = &X[((size_t)b * Kc + c) * TT];
        int st0 = n0 + bn + tap - ((R-1)/2);
        if (R == 1 || (st0 >= 0 && st0 + 3 < TT)) {
          f[j] = *(const v4fu*)&src[st0];
          if (MIN_) {
            const float* mk = &xmask[(size_t)b*TT + st0];
            f[j][0] *= mk[0]; f[j][1] *= mk[1]; f[j][2] *= mk[2]; f[j][3] *= mk[3];
          }
        } else {
          #pragma unroll
          for (int i = 0; i < 4; ++i) {
            int st = st0 + i;
            float v = 0.f;
            if (st >= 0 && st < TT) {
              v = src[st];
              if (MIN_) v *= xmask[(size_t)b*TT + st];
            }
            f[j][i] = v;
          }
        }
      }
      #pragma unroll
      for (int i = 0; i < 4; ++i) {
        v4fu col = { f[0][i], f[1][i], f[2][i], f[3][i] };
        *(v4bf*)&Bl[(bn+i)*LDK + bk] = pack4v(col);
      }
    }
    __syncthreads();
    v16bf af0 = frag_ld(&Al[(wm*32 + fr)*LDK], kh);
    v16bf af1 = frag_ld(&Al[(wm*32 + 16 + fr)*LDK], kh);
    v16bf bf0 = frag_ld(&Bl[(wn*32 + fr)*LDK], kh);
    v16bf bf1 = frag_ld(&Bl[(wn*32 + 16 + fr)*LDK], kh);
    a00 = wmma_bf(af0, bf0, a00);
    a01 = wmma_bf(af0, bf1, a01);
    a10 = wmma_bf(af1, bf0, a10);
    a11 = wmma_bf(af1, bf1, a11);
    __syncthreads();
  }
  const int moff = (lane >> 4) << 3;
  const int gn0 = n0 + wn*32 + fr;
  float mo0 = 1.f, mo1 = 1.f;
  if (MOUT) { mo0 = xmask[(size_t)b*TT + gn0]; mo1 = xmask[(size_t)b*TT + gn0 + 16]; }
  #pragma unroll
  for (int half = 0; half < 2; ++half) {
    const v8f& c0 = half ? a10 : a00;
    const v8f& c1 = half ? a11 : a01;
    #pragma unroll
    for (int r = 0; r < 8; ++r) {
      int gm = m0 + wm*32 + half*16 + moff + r;
      if (gm < M) {
        float bv = bias[gm];
        float v0 = (c0[r] + bv) * alpha;
        float v1 = (c1[r] + bv) * alpha;
        if (RELU) { v0 = fmaxf(v0, 0.f); v1 = fmaxf(v1, 0.f); }
        Y[((size_t)b*M + gm)*TT + gn0]      = v0 * mo0;
        Y[((size_t)b*M + gm)*TT + gn0 + 16] = v1 * mo1;
      }
    }
  }
}

// ---------------------------------------------------------------------------
// S[bh,t,s] = sum_d Q[bh,d,t] * K[bh,d,s]   (scale folded into Q)
__global__ __launch_bounds__(NT) void scores_kernel(
    const float* __restrict__ Q, const float* __restrict__ Kt, float* __restrict__ S)
{
  __shared__ alignas(16) bhalf Al[BMT*LDK];
  __shared__ alignas(16) bhalf Bl[BNT*LDK];
  const int bh = blockIdx.z;
  const int m0 = blockIdx.y * BMT;   // t rows
  const int n0 = blockIdx.x * BNT;   // s cols
  const int tid = threadIdx.x, lane = tid & 31, wid = tid >> 5;
  const int wm = wid & 1, wn = wid >> 1;
  const int fr = lane & 15, kh = (lane >> 4) << 3;
  v8f a00 = {0,0,0,0,0,0,0,0}, a01 = a00, a10 = a00, a11 = a00;

  for (int k0 = 0; k0 < DKH; k0 += BKT) {
    // B tile: 256 4x4 transpose blocks (global contiguous along s)
    {
      int bn = (tid & 31) * 4;
      int bk = (tid >> 5) * 4;
      v4fu f[4];
      #pragma unroll
      for (int j = 0; j < 4; ++j)
        f[j] = *(const v4fu*)&Kt[((size_t)bh*DKH + k0 + bk + j)*TT + (n0 + bn)];
      #pragma unroll
      for (int i = 0; i < 4; ++i) {
        v4fu col = { f[0][i], f[1][i], f[2][i], f[3][i] };
        *(v4bf*)&Bl[(bn+i)*LDK + bk] = pack4v(col);
      }
    }
    // A tile: 128 4x4 transpose blocks (global contiguous along t)
    if (tid < 128) {
      int bm = (tid & 15) * 4;
      int bk = (tid >> 4) * 4;
      v4fu f[4];
      #pragma unroll
      for (int j = 0; j < 4; ++j)
        f[j] = *(const v4fu*)&Q[((size_t)bh*DKH + k0 + bk + j)*TT + (m0 + bm)];
      #pragma unroll
      for (int i = 0; i < 4; ++i) {
        v4fu col = { f[0][i], f[1][i], f[2][i], f[3][i] };
        *(v4bf*)&Al[(bm+i)*LDK + bk] = pack4v(col);
      }
    }
    __syncthreads();
    v16bf af0 = frag_ld(&Al[(wm*32 + fr)*LDK], kh);
    v16bf af1 = frag_ld(&Al[(wm*32 + 16 + fr)*LDK], kh);
    v16bf bf0 = frag_ld(&Bl[(wn*32 + fr)*LDK], kh);
    v16bf bf1 = frag_ld(&Bl[(wn*32 + 16 + fr)*LDK], kh);
    a00 = wmma_bf(af0, bf0, a00);
    a01 = wmma_bf(af0, bf1, a01);
    a10 = wmma_bf(af1, bf0, a10);
    a11 = wmma_bf(af1, bf1, a11);
    __syncthreads();
  }
  const int moff = (lane >> 4) << 3;
  const int gn0 = n0 + wn*32 + fr;
  #pragma unroll
  for (int half = 0; half < 2; ++half) {
    const v8f& c0 = half ? a10 : a00;
    const v8f& c1 = half ? a11 : a01;
    #pragma unroll
    for (int r = 0; r < 8; ++r) {
      int gm = m0 + wm*32 + half*16 + moff + r;
      S[((size_t)bh*TT + gm)*TT + gn0]      = c0[r];
      S[((size_t)bh*TT + gm)*TT + gn0 + 16] = c1[r];
    }
  }
}

// ---------------------------------------------------------------------------
// O[bh,d,t] = sum_s V[bh,d,s] * P[bh,t,s]; P is bf16 -> async DMA straight into LDS.
__global__ __launch_bounds__(NT) void pv_kernel(
    const float* __restrict__ V, const bhalf* __restrict__ P, float* __restrict__ O)
{
  __shared__ alignas(16) bhalf Al[BMT*LDK];
  __shared__ alignas(16) bhalf Bl[BNT*LDK];
  const int bh = blockIdx.z;
  const int m0 = blockIdx.y * BMT;   // d rows
  const int n0 = blockIdx.x * BNT;   // t cols
  const int tid = threadIdx.x, lane = tid & 31, wid = tid >> 5;
  const int wm = wid & 1, wn = wid >> 1;
  const int fr = lane & 15, kh = (lane >> 4) << 3;
  v8f a00 = {0,0,0,0,0,0,0,0}, a01 = a00, a10 = a00, a11 = a00;

  for (int k0 = 0; k0 < TT; k0 += BKT) {
    // B tile: 128 rows x 32 k of bf16, global->LDS with no VGPR round-trip (ASYNCcnt)
    #pragma unroll
    for (int it = 0; it < (BNT*BKT)/(8*NT); ++it) {
      int g = tid + it*NT;            // 512 b128 segments
      int n   = g >> 2;
      int seg = (g & 3) * 8;
      unsigned dst = lds_addr(&Bl[n*LDK + seg]);
      const void* src = (const void*)&P[((size_t)bh*TT + n0 + n)*TT + (k0 + seg)];
      asm volatile("global_load_async_to_lds_b128 %0, %1, off" :: "v"(dst), "v"(src) : "memory");
    }
    // A tile: f32 V, contiguous k: b128 load + b64 packed store
    #pragma unroll
    for (int it = 0; it < (BMT*BKT)/(4*NT); ++it) {
      int g = tid + it*NT;
      int k4 = (g & 7) * 4;
      int m  = g >> 3;
      int gm = m0 + m;
      v4fu w = {0,0,0,0};
      if (gm < DKH) w = *(const v4fu*)&V[((size_t)bh*DKH + gm)*TT + k0 + k4];
      *(v4bf*)&Al[m*LDK + k4] = pack4v(w);
    }
    asm volatile("s_wait_asynccnt 0x0" ::: "memory");
    __syncthreads();
    v16bf af0 = frag_ld(&Al[(wm*32 + fr)*LDK], kh);
    v16bf af1 = frag_ld(&Al[(wm*32 + 16 + fr)*LDK], kh);
    v16bf bf0 = frag_ld(&Bl[(wn*32 + fr)*LDK], kh);
    v16bf bf1 = frag_ld(&Bl[(wn*32 + 16 + fr)*LDK], kh);
    a00 = wmma_bf(af0, bf0, a00);
    a01 = wmma_bf(af0, bf1, a01);
    a10 = wmma_bf(af1, bf0, a10);
    a11 = wmma_bf(af1, bf1, a11);
    __syncthreads();
  }
  const int moff = (lane >> 4) << 3;
  const int gn0 = n0 + wn*32 + fr;
  #pragma unroll
  for (int half = 0; half < 2; ++half) {
    const v8f& c0 = half ? a10 : a00;
    const v8f& c1 = half ? a11 : a01;
    #pragma unroll
    for (int r = 0; r < 8; ++r) {
      int gm = m0 + wm*32 + half*16 + moff + r;
      if (gm < DKH) {
        O[((size_t)bh*DKH + gm)*TT + gn0]      = c0[r];
        O[((size_t)bh*DKH + gm)*TT + gn0 + 16] = c1[r];
      }
    }
  }
}

// ---------------------------------------------------------------------------
// rb[bh,t,r] = sum_d Qs[bh,d,t] * erk[r,d]  (banded rel-K logits, r = s-t+4)
__global__ void relbias_kernel(const float* __restrict__ Q,
                               const float* __restrict__ erk, float* __restrict__ rb)
{
  int idx = blockIdx.x*blockDim.x + threadIdx.x;
  if (idx >= BHD*TT*9) return;
  int r = idx % 9;
  int t = (idx / 9) & (TT-1);
  int bh = idx / (9*TT);
  const float* qp = Q + (size_t)bh*DKH*TT + t;
  const float* ep = erk + (size_t)r*DKH;
  float s = 0.f;
  #pragma unroll 4
  for (int d = 0; d < DKH; ++d) s += qp[(size_t)d*TT] * ep[d];
  rb[idx] = s;
}

// ---------------------------------------------------------------------------
// Row softmax with banded rel bias + mask; writes P in bf16.
__global__ __launch_bounds__(256) void softmax_kernel(
    const float* __restrict__ S, const float* __restrict__ rb,
    const float* __restrict__ xmask, bhalf* __restrict__ P)
{
  const int row = blockIdx.x;          // bh*T + t
  const int t = row & (TT-1);
  const int bh = row >> 10;
  const int b = bh / NH;
  const int tid = threadIdx.x, lane = tid & 31, wid = tid >> 5;
  __shared__ float sh[8];
  const float mt = xmask[(size_t)b*TT + t];
  float vals[TT/256];
  float mx = -3.0e38f;
  #pragma unroll
  for (int i = 0; i < TT/256; ++i) {
    int s = tid + i*256;
    float v = S[(size_t)row*TT + s];
    int d = s - t;
    if (d >= -WSZ && d <= WSZ) v += rb[(size_t)row*9 + d + WSZ];
    if (mt == 0.f || xmask[(size_t)b*TT + s] == 0.f) v = -1e4f;
    vals[i] = v;
    mx = fmaxf(mx, v);
  }
  for (int o = 16; o; o >>= 1) mx = fmaxf(mx, __shfl_xor(mx, o, 32));
  if (lane == 0) sh[wid] = mx;
  __syncthreads();
  mx = sh[0];
  #pragma unroll
  for (int i = 1; i < 8; ++i) mx = fmaxf(mx, sh[i]);
  __syncthreads();
  float ls = 0.f;
  #pragma unroll
  for (int i = 0; i < TT/256; ++i) { vals[i] = __expf(vals[i] - mx); ls += vals[i]; }
  for (int o = 16; o; o >>= 1) ls += __shfl_xor(ls, o, 32);
  if (lane == 0) sh[wid] = ls;
  __syncthreads();
  float tot = 0.f;
  #pragma unroll
  for (int i = 0; i < 8; ++i) tot += sh[i];
  float inv = 1.f / tot;
  #pragma unroll
  for (int i = 0; i < TT/256; ++i)
    P[(size_t)row*TT + tid + i*256] = f2bf(vals[i] * inv);
}

// ---------------------------------------------------------------------------
// O[bh,d,t] += sum_{r=0..8} P[bh,t,t+r-4] * erv[r,d]
__global__ void relv_kernel(const bhalf* __restrict__ P,
                            const float* __restrict__ erv, float* __restrict__ O)
{
  int idx = blockIdx.x*blockDim.x + threadIdx.x;
  if (idx >= BHD*DKH*TT) return;
  int t = idx & (TT-1);
  int d = (idx / TT) % DKH;
  int bh = idx / (TT*DKH);
  const bhalf* pr = P + ((size_t)bh*TT + t)*TT;
  float s = 0.f;
  #pragma unroll
  for (int r = 0; r < 9; ++r) {
    int sc = t + r - WSZ;
    if (sc >= 0 && sc < TT) s += bf2f(pr[sc]) * erv[r*DKH + d];
  }
  O[((size_t)bh*DKH + d)*TT + t] += s;
}

// ---------------------------------------------------------------------------
// x = LayerNorm_channel(x + y) * g + b   (192 threads = 6 waves per (b,t))
__global__ __launch_bounds__(C_) void add_ln_kernel(
    float* __restrict__ x, const float* __restrict__ y,
    const float* __restrict__ g, const float* __restrict__ be)
{
  const int blk = blockIdx.x;
  const int t = blk & (TT-1);
  const int b = blk >> 10;
  const int c = threadIdx.x;
  const int lane = c & 31, wid = c >> 5;
  __shared__ float sh[8];
  size_t off = ((size_t)b*C_ + c)*TT + t;
  float v = x[off] + y[off];
  float s = v;
  for (int o = 16; o; o >>= 1) s += __shfl_xor(s, o, 32);
  if (lane == 0) sh[wid] = s;
  __syncthreads();
  float mean = 0.f;
  #pragma unroll
  for (int i = 0; i < 6; ++i) mean += sh[i];
  mean *= (1.f/C_);
  __syncthreads();
  float d = v - mean;
  float q = d*d;
  for (int o = 16; o; o >>= 1) q += __shfl_xor(q, o, 32);
  if (lane == 0) sh[wid] = q;
  __syncthreads();
  float var = 0.f;
  #pragma unroll
  for (int i = 0; i < 6; ++i) var += sh[i];
  var *= (1.f/C_);
  x[off] = d * rsqrtf(var + 1e-5f) * g[c] + be[c];
}

__global__ void mask_copy_kernel(const float* __restrict__ in,
                                 const float* __restrict__ xmask,
                                 float* __restrict__ out)
{
  int idx = blockIdx.x*blockDim.x + threadIdx.x;
  if (idx >= BB*C_*TT) return;
  int t = idx & (TT-1);
  int b = idx / (C_*TT);
  out[idx] = in[idx] * xmask[(size_t)b*TT + t];
}

// ---------------------------------------------------------------------------
extern "C" void kernel_launch(void* const* d_in, const int* in_sizes, int n_in,
                              void* d_out, int out_size, void* d_ws, size_t ws_size,
                              hipStream_t stream)
{
  (void)in_sizes; (void)n_in; (void)out_size; (void)ws_size;
  const float* x_in  = (const float*)d_in[0];
  const float* xmask = (const float*)d_in[1];
  const float* Wq = (const float*)d_in[2];  const float* bq = (const float*)d_in[3];
  const float* Wk = (const float*)d_in[4];  const float* bk = (const float*)d_in[5];
  const float* Wv = (const float*)d_in[6];  const float* bv = (const float*)d_in[7];
  const float* Wo = (const float*)d_in[8];  const float* bo = (const float*)d_in[9];
  const float* erk = (const float*)d_in[10];
  const float* erv = (const float*)d_in[11];
  const float* ln1g = (const float*)d_in[12]; const float* ln1b = (const float*)d_in[13];
  const float* w1 = (const float*)d_in[14];   const float* b1 = (const float*)d_in[15];
  const float* w2 = (const float*)d_in[16];   const float* b2 = (const float*)d_in[17];
  const float* ln2g = (const float*)d_in[18]; const float* ln2b = (const float*)d_in[19];
  float* out = (float*)d_out;

  char* ws = (char*)d_ws;
  const size_t NX  = (size_t)BB*C_*TT;
  const size_t SZX = NX*sizeof(float);
  const size_t SZS = (size_t)BHD*TT*TT*sizeof(float);
  const size_t SZP = (size_t)BHD*TT*TT*sizeof(bhalf);
  float* x  = (float*)(ws);
  float* y  = (float*)(ws + SZX);
  float* Qb = (float*)(ws + 2*SZX);
  float* Kb = (float*)(ws + 3*SZX);
  float* Vb = (float*)(ws + 4*SZX);
  float* S  = (float*)(ws + 5*SZX);
  bhalf* P  = (bhalf*)(ws + 5*SZX + SZS);
  float* rb = (float*)(ws + 5*SZX + SZS + SZP);
  float* hb = S;    // FFN hidden reuses scores region (dead after softmax)
  float* O  = Qb;   // attention output reuses Q (dead after relbias/softmax)

  const float scale = 1.0f / sqrtf((float)DKH);
  dim3 blk(NT);
  dim3 gproj(TT/BNT, C_/BMT, BB);

  mask_copy_kernel<<<dim3((unsigned)((NX+255)/256)), dim3(256), 0, stream>>>(x_in, xmask, x);

  for (int l = 0; l < NL; ++l) {
    const float* Wq_l = Wq + (size_t)l*C_*C_;   const float* bq_l = bq + (size_t)l*C_;
    const float* Wk_l = Wk + (size_t)l*C_*C_;   const float* bk_l = bk + (size_t)l*C_;
    const float* Wv_l = Wv + (size_t)l*C_*C_;   const float* bv_l = bv + (size_t)l*C_;
    const float* Wo_l = Wo + (size_t)l*C_*C_;   const float* bo_l = bo + (size_t)l*C_;
    const float* erk_l = erk + (size_t)l*(2*WSZ+1)*DKH;
    const float* erv_l = erv + (size_t)l*(2*WSZ+1)*DKH;
    const float* w1_l = w1 + (size_t)l*FC_*C_*3; const float* b1_l = b1 + (size_t)l*FC_;
    const float* w2_l = w2 + (size_t)l*C_*FC_*3; const float* b2_l = b2 + (size_t)l*C_;

    gemm_conv_kernel<1,false,false,false><<<gproj, blk, 0, stream>>>(Wq_l, bq_l, x, xmask, Qb, C_, C_, scale);
    gemm_conv_kernel<1,false,false,false><<<gproj, blk, 0, stream>>>(Wk_l, bk_l, x, xmask, Kb, C_, C_, 1.f);
    gemm_conv_kernel<1,false,false,false><<<gproj, blk, 0, stream>>>(Wv_l, bv_l, x, xmask, Vb, C_, C_, 1.f);

    scores_kernel<<<dim3(TT/BNT, TT/BMT, BHD), blk, 0, stream>>>(Qb, Kb, S);
    relbias_kernel<<<dim3((BHD*TT*9 + 255)/256), dim3(256), 0, stream>>>(Qb, erk_l, rb);
    softmax_kernel<<<dim3(BHD*TT), dim3(256), 0, stream>>>(S, rb, xmask, P);
    pv_kernel<<<dim3(TT/BNT, (DKH + BMT - 1)/BMT, BHD), blk, 0, stream>>>(Vb, P, O);
    relv_kernel<<<dim3((BHD*DKH*TT + 255)/256), dim3(256), 0, stream>>>(P, erv_l, O);

    gemm_conv_kernel<1,false,false,false><<<gproj, blk, 0, stream>>>(Wo_l, bo_l, O, xmask, y, C_, C_, 1.f);
    add_ln_kernel<<<dim3(BB*TT), dim3(C_), 0, stream>>>(x, y, ln1g + (size_t)l*C_, ln1b + (size_t)l*C_);

    gemm_conv_kernel<3,true, true, false><<<dim3(TT/BNT, FC_/BMT, BB), blk, 0, stream>>>(w1_l, b1_l, x,  xmask, hb, FC_, C_, 1.f);
    gemm_conv_kernel<3,false,true, true ><<<dim3(TT/BNT, C_/BMT,  BB), blk, 0, stream>>>(w2_l, b2_l, hb, xmask, y,  C_, FC_, 1.f);
    add_ln_kernel<<<dim3(BB*TT), dim3(C_), 0, stream>>>(x, y, ln2g + (size_t)l*C_, ln2b + (size_t)l*C_);
  }

  mask_copy_kernel<<<dim3((unsigned)((NX+255)/256)), dim3(256), 0, stream>>>(x, xmask, out);
}
